// ContinuousGenHyperConnections_66941360275518
// MI455X (gfx1250) — matrix operand order repaired and checked by reading, not compiled, not run on hardware
//
#include <hip/hip_runtime.h>
#include <hip/hip_bf16.h>

typedef __attribute__((ext_vector_type(16))) _Float16 v16h;
typedef __attribute__((ext_vector_type(8)))  float    v8f;

#define BATCH   8192
#define NSTR    4
#define EMBD    2048
#define IND     8192   // NSTR * EMBD

// ---------------------------------------------------------------------------
// CDNA5 async memory->LDS staging (ASYNCcnt-tracked), with graceful fallback.
// Builtin signature (from hipcc diagnostic): param0 = int4-vector ptr in AS1
// (printed as "__device__"), param1 = int4-vector ptr in AS3, then two imm ints.
// ---------------------------------------------------------------------------
#if defined(__AMDGCN__) && __has_builtin(__builtin_amdgcn_global_load_async_to_lds_b128)
#define ASYNC_LDS 1
#else
#define ASYNC_LDS 0
#endif

#if defined(__AMDGCN__)
typedef int v4i_vs __attribute__((vector_size(16)));
typedef __attribute__((address_space(1))) v4i_vs as1_v4i;
typedef __attribute__((address_space(3))) v4i_vs as3_v4i;
#endif

__device__ __forceinline__ void copy16_to_lds(void* lds_dst, const void* gsrc) {
#if ASYNC_LDS
    // generic->AS1 is an identity on AMDGPU; generic LDS address truncated to
    // 32 bits is the LDS byte offset (ISA 10.2 aperture rule).
    __builtin_amdgcn_global_load_async_to_lds_b128(
        (as1_v4i*)(uintptr_t)gsrc,
        (as3_v4i*)(uint32_t)(uintptr_t)lds_dst,
        0, 0);
#else
    *(float4*)lds_dst = *(const float4*)gsrc;
#endif
}

__device__ __forceinline__ void wait_async_copies() {
#if ASYNC_LDS
#if __has_builtin(__builtin_amdgcn_s_wait_asynccnt)
    __builtin_amdgcn_s_wait_asynccnt(0);
#else
    asm volatile("s_wait_asynccnt 0x0" ::: "memory");
#endif
#endif
}

__device__ __forceinline__ float sigmoidf_(float z) {
    return 1.0f / (1.0f + __expf(-z));
}

// ---------------------------------------------------------------------------
// Kernel 0: W_mod f32 -> f16 (row-major [d, k], 2048x2048). 8 MB, L2-resident.
// ---------------------------------------------------------------------------
__global__ void convert_wmod_kernel(const float* __restrict__ w,
                                    _Float16* __restrict__ o, int n) {
    for (int i = blockIdx.x * blockDim.x + threadIdx.x; i < n;
         i += gridDim.x * blockDim.x) {
        o[i] = (_Float16)w[i];
    }
}

// ---------------------------------------------------------------------------
// Kernel 1: per-sample fused front end (RMS stats, 42 projections, 4x4 expm,
// mixed = E @ x -> d_out residual base, branch -> f16 ws, ww -> ws).
// One 256-thread block per sample.
// ---------------------------------------------------------------------------
__global__ __launch_bounds__(256) void fused_front_kernel(
    const float* __restrict__ x,
    const float* __restrict__ read_in,  const float* __restrict__ alpha_read_in,
    const float* __restrict__ write_out,const float* __restrict__ alpha_write_out,
    const float* __restrict__ W_read,   const float* __restrict__ W_write,
    const float* __restrict__ log_dt_c, const float* __restrict__ log_dt_d,
    const float* __restrict__ W_dtc,    const float* __restrict__ b_dtc,
    const float* __restrict__ W_dtd,    const float* __restrict__ b_dtd,
    const float* __restrict__ conserv_A,const float* __restrict__ W_conv,
    const float* __restrict__ b_conv,
    const float* __restrict__ diss_A,   const float* __restrict__ W_diss,
    const float* __restrict__ b_diss,
    float* __restrict__ out,            // [B, 4, 2048] residual base (mixed)
    _Float16* __restrict__ branch16,    // [B, 2048]
    float* __restrict__ ww_out)         // [B, 4]
{
    const int b = blockIdx.x;
    const int t = threadIdx.x;

    __shared__ float s_proj[42];
    __shared__ float s_ssq;
    __shared__ float s_E[16];
    __shared__ float s_rw[4];

    if (t < 42) s_proj[t] = 0.0f;
    if (t == 63) s_ssq = 0.0f;
    __syncthreads();

    const float* xb = x + (size_t)b * IND;

    float xc[32];
    float ss = 0.0f;
#pragma unroll
    for (int i = 0; i < 32; ++i) {
        xc[i] = xb[t * 32 + i];
        ss += xc[i] * xc[i];
    }
#pragma unroll
    for (int off = 16; off; off >>= 1) ss += __shfl_down(ss, off, 32);
    if ((t & 31) == 0) atomicAdd(&s_ssq, ss);
    __syncthreads();

    const float inv = rsqrtf(s_ssq * (1.0f / (float)IND) + 1e-6f);

    for (int r = 0; r < 42; ++r) {
        const float* wb;
        if (r < 16)       wb = W_conv  + (size_t)r * IND;
        else if (r < 32)  wb = W_diss  + (size_t)(r - 16) * IND;
        else if (r == 32) wb = W_dtc;
        else if (r == 33) wb = W_dtd;
        else if (r < 38)  wb = W_read  + (size_t)(r - 34) * IND;
        else              wb = W_write + (size_t)(r - 38) * IND;
        wb += t * 32;
        float d = 0.0f;
#pragma unroll
        for (int i = 0; i < 32; ++i) d += xc[i] * wb[i];
#pragma unroll
        for (int off = 16; off; off >>= 1) d += __shfl_down(d, off, 32);
        if ((t & 31) == 0) atomicAdd(&s_proj[r], d);
    }
    __syncthreads();
    if (t < 42) s_proj[t] *= inv;   // dot(xn, w) = inv * dot(x, w)
    __syncthreads();

    if (t == 0) {
        const float DTmn = 1e-3f, DTmx = 1.0f;
        const float dtc = DTmn + (DTmx - DTmn) *
                          sigmoidf_(log_dt_c[0] + s_proj[32] + b_dtc[0]);
        const float dtd = DTmn + (DTmx - DTmn) *
                          sigmoidf_(log_dt_d[0] + s_proj[33] + b_dtd[0]);

        float Mg[4][4], Rg[4][4], Ag[4][4];
        for (int i = 0; i < 4; ++i)
            for (int j = 0; j < 4; ++j) {
                Mg[i][j] = conserv_A[i * 4 + j] + s_proj[i * 4 + j] + b_conv[i * 4 + j];
                Rg[i][j] = diss_A[i * 4 + j] + s_proj[16 + i * 4 + j] + b_diss[i * 4 + j];
            }
        for (int i = 0; i < 4; ++i)
            for (int j = 0; j < 4; ++j) {
                float k = 0.0f;
                for (int c = 0; c < 4; ++c) k += Rg[i][c] * Rg[j][c];
                Ag[i][j] = dtc * (Mg[i][j] - Mg[j][i]) - dtd * k;
            }

        // expm: scale by 2^-4, order-8 Taylor, square 4 times
        float E[4][4], T[4][4], Tn[4][4];
        for (int i = 0; i < 4; ++i)
            for (int j = 0; j < 4; ++j) {
                Ag[i][j] *= 0.0625f;
                E[i][j] = ((i == j) ? 1.0f : 0.0f) + Ag[i][j];
                T[i][j] = Ag[i][j];
            }
        for (int k = 2; k <= 8; ++k) {
            const float rk = 1.0f / (float)k;
            for (int i = 0; i < 4; ++i)
                for (int j = 0; j < 4; ++j) {
                    float s = 0.0f;
                    for (int c = 0; c < 4; ++c) s += T[i][c] * Ag[c][j];
                    Tn[i][j] = s * rk;
                }
            for (int i = 0; i < 4; ++i)
                for (int j = 0; j < 4; ++j) {
                    T[i][j] = Tn[i][j];
                    E[i][j] += T[i][j];
                }
        }
        for (int sq = 0; sq < 4; ++sq) {
            for (int i = 0; i < 4; ++i)
                for (int j = 0; j < 4; ++j) {
                    float s = 0.0f;
                    for (int c = 0; c < 4; ++c) s += E[i][c] * E[c][j];
                    Tn[i][j] = s;
                }
            for (int i = 0; i < 4; ++i)
                for (int j = 0; j < 4; ++j) E[i][j] = Tn[i][j];
        }
        for (int i = 0; i < 16; ++i) s_E[i] = E[i / 4][i % 4];

        for (int n = 0; n < 4; ++n) {
            s_rw[n] = sigmoidf_(read_in[n] + alpha_read_in[0] * s_proj[34 + n]);
            ww_out[(size_t)b * 4 + n] =
                write_out[n] + alpha_write_out[0] * s_proj[38 + n];
        }
    }
    __syncthreads();

    // mixed = E @ x ; branch = rw . mixed
#pragma unroll
    for (int q = 0; q < 8; ++q) {
        const int d = t + 256 * q;
        float xj[4];
#pragma unroll
        for (int j = 0; j < 4; ++j) xj[j] = xb[j * EMBD + d];
        float br = 0.0f;
#pragma unroll
        for (int n = 0; n < 4; ++n) {
            float mn = 0.0f;
#pragma unroll
            for (int j = 0; j < 4; ++j) mn += s_E[n * 4 + j] * xj[j];
            out[(size_t)b * IND + n * EMBD + d] = mn;
            br += s_rw[n] * mn;
        }
        branch16[(size_t)b * EMBD + d] = (_Float16)br;
    }
}

// ---------------------------------------------------------------------------
// Kernel 2: y = branch @ W_mod.T via v_wmma_f32_16x16x32_f16, double-buffered
// LDS with async global->LDS staging (ASYNCcnt pipeline): the copy for K-step
// kk+32 overlaps the two WMMAs of step kk. Epilogue fuses
//   out[b, n, d] += ww[b, n] * y[b, d]
// Block tile 64 (batch) x 64 (d); 8 waves, each wave owns one 16-row band and
// two 16-col WMMA tiles.
// ---------------------------------------------------------------------------
__global__ __launch_bounds__(256) void gemm_writeout_kernel(
    const _Float16* __restrict__ A16,   // [B, 2048] branch
    const _Float16* __restrict__ Bw16,  // [2048, 2048] W_mod row-major [d, k]
    const float* __restrict__ ww,       // [B, 4]
    float* __restrict__ out)            // [B, 4, 2048]
{
    __shared__ alignas(32) _Float16 sA[2][64 * 32];   // [buf][row][k]
    __shared__ alignas(32) _Float16 sB[2][64 * 32];   // [buf][d][k]

    const int t    = threadIdx.x;
    const int lane = t & 31;
    const int wave = t >> 5;
    const int g    = lane >> 4;      // half-wave group
    const int m    = lane & 15;
    const int tr   = wave & 3;       // 16-row band within block tile
    const int c0   = (wave >> 2) * 32;
    const int rowBase = blockIdx.y * 64;
    const int colBase = blockIdx.x * 64;

    const int ldRow   = t >> 2;
    const int ldChunk = (t & 3) * 8;
    const int ldsOff  = ldRow * 32 + ldChunk;
    const size_t aOff = (size_t)(rowBase + ldRow) * EMBD + ldChunk;
    const size_t bOff = (size_t)(colBase + ldRow) * EMBD + ldChunk;

    v8f acc0 = {};
    v8f acc1 = {};

    // Prologue: stage K-step 0 into buffer 0.
    copy16_to_lds(&sA[0][ldsOff], &A16[aOff]);
    copy16_to_lds(&sB[0][ldsOff], &Bw16[bOff]);
    wait_async_copies();
    __syncthreads();

    for (int kk = 0; kk < EMBD; kk += 32) {
        const int cur = (kk >> 5) & 1;
        const int nxt = cur ^ 1;

        // Issue async staging for the next K-step; overlaps with WMMA below.
        if (kk + 32 < EMBD) {
            copy16_to_lds(&sA[nxt][ldsOff], &A16[aOff + kk + 32]);
            copy16_to_lds(&sB[nxt][ldsOff], &Bw16[bOff + kk + 32]);
#if !ASYNC_LDS
            if (kk + 64 < EMBD)
                __builtin_prefetch(&Bw16[bOff + kk + 64], 0, 3);
#endif
        }

        // A fragment: lane (g, m) holds row M=m, K = 16g..16g+15 contiguous
        // (matches B fragment K order).
        v16h a;
        const _Float16* arow = &sA[cur][(tr * 16 + m) * 32];
#pragma unroll
        for (int vv = 0; vv < 8; ++vv) {
            const int kb = 16 * g + 2 * vv;
            a[2 * vv]     = arow[kb];
            a[2 * vv + 1] = arow[kb + 1];
        }

        // B fragments: lane (g, m) holds column N=m, K = 16g..16g+15 contiguous.
        v16h b0 = *(const v16h*)&sB[cur][(c0 + m) * 32 + 16 * g];
        v16h b1 = *(const v16h*)&sB[cur][(c0 + 16 + m) * 32 + 16 * g];

        acc0 = __builtin_amdgcn_wmma_f32_16x16x32_f16(
            false, a, false, b0, (short)0, acc0, false, false);
        acc1 = __builtin_amdgcn_wmma_f32_16x16x32_f16(
            false, a, false, b1, (short)0, acc1, false, false);

        // Publish next buffer: my async copies done, then block-wide barrier.
        wait_async_copies();
        __syncthreads();
    }

    // Epilogue: out[b, n, d] += ww[b, n] * y[b, d]
#pragma unroll
    for (int tile = 0; tile < 2; ++tile) {
        v8f acc = tile ? acc1 : acc0;
        const int gcol = colBase + c0 + tile * 16 + m;
#pragma unroll
        for (int r = 0; r < 8; ++r) {
            const int grow = rowBase + tr * 16 + 8 * g + r;  // C layout: M = r + 8g
            const float y = acc[r];
#pragma unroll
            for (int n = 0; n < 4; ++n) {
                const size_t oi = (((size_t)grow * 4 + n) << 11) + gcol;
                out[oi] += ww[(size_t)grow * 4 + n] * y;
            }
        }
    }
}

// ---------------------------------------------------------------------------
// Launch. Workspace layout:
//   [0, 8 MB)        W_mod f16
//   [8 MB, 40 MB)    branch f16 [B, 2048]
//   [40 MB, +128 KB) ww f32 [B, 4]
// ---------------------------------------------------------------------------
extern "C" void kernel_launch(void* const* d_in, const int* in_sizes, int n_in,
                              void* d_out, int out_size, void* d_ws, size_t ws_size,
                              hipStream_t stream) {
    (void)in_sizes; (void)n_in; (void)out_size; (void)ws_size;

    const float* x              = (const float*)d_in[0];
    const float* read_in        = (const float*)d_in[1];
    const float* alpha_read_in  = (const float*)d_in[2];
    const float* write_out      = (const float*)d_in[3];
    const float* alpha_write_out= (const float*)d_in[4];
    const float* W_read         = (const float*)d_in[5];
    const float* W_write        = (const float*)d_in[6];
    const float* log_dt_c       = (const float*)d_in[7];
    const float* log_dt_d       = (const float*)d_in[8];
    const float* W_dtc          = (const float*)d_in[9];
    const float* b_dtc          = (const float*)d_in[10];
    const float* W_dtd          = (const float*)d_in[11];
    const float* b_dtd          = (const float*)d_in[12];
    const float* conserv_A      = (const float*)d_in[13];
    const float* W_conv         = (const float*)d_in[14];
    const float* b_conv         = (const float*)d_in[15];
    const float* diss_A         = (const float*)d_in[16];
    const float* W_diss         = (const float*)d_in[17];
    const float* b_diss         = (const float*)d_in[18];
    const float* W_mod          = (const float*)d_in[19];

    _Float16* wmod16   = (_Float16*)d_ws;
    _Float16* branch16 = wmod16 + (size_t)EMBD * EMBD;
    float*    ww       = (float*)(branch16 + (size_t)BATCH * EMBD);
    float*    out      = (float*)d_out;

    convert_wmod_kernel<<<4096, 256, 0, stream>>>(W_mod, wmod16, EMBD * EMBD);

    fused_front_kernel<<<BATCH, 256, 0, stream>>>(
        x, read_in, alpha_read_in, write_out, alpha_write_out,
        W_read, W_write, log_dt_c, log_dt_d, W_dtc, b_dtc, W_dtd, b_dtd,
        conserv_A, W_conv, b_conv, diss_A, W_diss, b_diss,
        out, branch16, ww);

    gemm_writeout_kernel<<<dim3(EMBD / 64, BATCH / 64), 256, 0, stream>>>(
        branch16, wmod16, ww, out);
}
